// Linear1d_4569845203007
// MI455X (gfx1250) — matrix operand order repaired and checked by reading, not compiled, not run on hardware
//
#include <hip/hip_runtime.h>

// out[n,o] = (1/sqrt(2048)) * sum_i X[n,i]*W[o,i] + 0.1*b[o]
// Phase 1: decompose fp32 X, W into bf16 hi/lo planes (in d_ws).
// Phase 2: 3-term bf16 WMMA GEMM (hi*hi + hi*lo + lo*hi ~ fp32 accuracy),
//          staging tiles with async global->LDS copies, double-buffered LDS.

#define NROWS 16384
#define DIN   2048
#define DOUT  2048

#define BM 128
#define BN 128
#define BK 32
#define LDSK 40   // padded LDS row stride (bf16 elems): 80B rows, conflict-free, 16B aligned

typedef __attribute__((ext_vector_type(16))) __bf16 bf16x16;
typedef __attribute__((ext_vector_type(8)))  float  floatx8;

struct Pack32 { uint4 a, b; };   // 32B = one per-lane bf16 WMMA fragment

__device__ __forceinline__ unsigned short f32_to_bf16_rne(float f) {
  unsigned int u = __float_as_uint(f);
  u += 0x7FFFu + ((u >> 16) & 1u);
  return (unsigned short)(u >> 16);
}
__device__ __forceinline__ float bf16_to_f32(unsigned short h) {
  return __uint_as_float(((unsigned int)h) << 16);
}
// Low 32 bits of a generic pointer to LDS == wave-relative LDS byte address (ISA 10.2).
__device__ __forceinline__ unsigned lds_off(const void* p) {
  return (unsigned)(unsigned long long)(uintptr_t)p;
}
// Async 16B-per-lane copy global -> LDS (GVS mode), tracked by ASYNCcnt.
__device__ __forceinline__ void async_copy_b128(unsigned lds_byte, unsigned voff,
                                                const void* sbase) {
  asm volatile("global_load_async_to_lds_b128 %0, %1, %2"
               :: "v"(lds_byte), "v"(voff), "s"(sbase) : "memory");
}
__device__ __forceinline__ void wait_async0() {
  asm volatile("s_wait_asynccnt 0x0" ::: "memory");
}

// ---------------- Phase 1: fp32 -> bf16 hi/lo planes ----------------
__global__ __launch_bounds__(256)
void split_planes(const float* __restrict__ src,
                  unsigned short* __restrict__ hi,
                  unsigned short* __restrict__ lo, int n4) {
  const int i = blockIdx.x * 256 + threadIdx.x;   // float4 index
  if (i >= n4) return;
  const float4 v = ((const float4*)src)[i];
  const float f[4] = {v.x, v.y, v.z, v.w};
  unsigned short h[4], l[4];
#pragma unroll
  for (int e = 0; e < 4; ++e) {
    h[e] = f32_to_bf16_rne(f[e]);
    l[e] = f32_to_bf16_rne(f[e] - bf16_to_f32(h[e]));
  }
  ushort4 hv; hv.x = h[0]; hv.y = h[1]; hv.z = h[2]; hv.w = h[3];
  ushort4 lv; lv.x = l[0]; lv.y = l[1]; lv.z = l[2]; lv.w = l[3];
  ((ushort4*)hi)[i] = hv;
  ((ushort4*)lo)[i] = lv;
}

// ---------------- Phase 2: bf16x3 WMMA GEMM with async staging ----------------
__global__ __launch_bounds__(256)
void gemm_planes_wmma(const unsigned short* __restrict__ Xhi,
                      const unsigned short* __restrict__ Xlo,
                      const unsigned short* __restrict__ Whi,
                      const unsigned short* __restrict__ Wlo,
                      const float* __restrict__ Bv,
                      float* __restrict__ O) {
  __shared__ __align__(16) unsigned short sAhi[2][BM * LDSK];
  __shared__ __align__(16) unsigned short sAlo[2][BM * LDSK];
  __shared__ __align__(16) unsigned short sBhi[2][BN * LDSK];
  __shared__ __align__(16) unsigned short sBlo[2][BN * LDSK];

  const int tid   = threadIdx.x;
  const int lane  = tid & 31;
  const int wave  = tid >> 5;
  const int wm    = wave >> 1;   // 0..3 : wave M group (2 tiles)
  const int wn    = wave & 1;    // 0..1 : wave N group (4 tiles)
  const int mrow  = lane & 15;
  const int khalf = lane >> 4;

  const int blockM = blockIdx.y * BM;
  const int blockN = blockIdx.x * BN;

  // Per-thread copy slots: chunk c = tid + p*256; 4 x 16B chunks per 32-col row.
  unsigned voffX[2], voffW[2], ldsIdx[2];
#pragma unroll
  for (int p = 0; p < 2; ++p) {
    const int c = tid + p * 256;
    const int row = c >> 2, ch = c & 3;
    voffX[p]  = (unsigned)((((blockM + row) * DIN) + ch * 8) * 2);  // bytes
    voffW[p]  = (unsigned)((((blockN + row) * DIN) + ch * 8) * 2);  // bytes
    ldsIdx[p] = (unsigned)(row * LDSK + ch * 8);                    // ushort elems
  }

  floatx8 acc[2][4];
#pragma unroll
  for (int i = 0; i < 2; ++i)
#pragma unroll
    for (int j = 0; j < 4; ++j) acc[i][j] = (floatx8)0.0f;

  // Prologue: fill buffer 0 with k=0 tile.
#pragma unroll
  for (int p = 0; p < 2; ++p) {
    async_copy_b128(lds_off(&sAhi[0][ldsIdx[p]]), voffX[p], Xhi);
    async_copy_b128(lds_off(&sAlo[0][ldsIdx[p]]), voffX[p], Xlo);
    async_copy_b128(lds_off(&sBhi[0][ldsIdx[p]]), voffW[p], Whi);
    async_copy_b128(lds_off(&sBlo[0][ldsIdx[p]]), voffW[p], Wlo);
  }
  wait_async0();
  __syncthreads();

  const int KT = DIN / BK;   // 64
  for (int kt = 0; kt < KT; ++kt) {
    const int buf = kt & 1;

    // Issue async copies for next k-step into the other buffer (overlaps WMMA).
    if (kt + 1 < KT) {
      const int nb = buf ^ 1;
      const int kn = (kt + 1) * BK;
#pragma unroll
      for (int p = 0; p < 2; ++p) {
        async_copy_b128(lds_off(&sAhi[nb][ldsIdx[p]]), voffX[p], Xhi + kn);
        async_copy_b128(lds_off(&sAlo[nb][ldsIdx[p]]), voffX[p], Xlo + kn);
        async_copy_b128(lds_off(&sBhi[nb][ldsIdx[p]]), voffW[p], Whi + kn);
        async_copy_b128(lds_off(&sBlo[nb][ldsIdx[p]]), voffW[p], Wlo + kn);
      }
    }

    const unsigned short* pAhi = sAhi[buf];
    const unsigned short* pAlo = sAlo[buf];
    const unsigned short* pBhi = sBhi[buf];
    const unsigned short* pBlo = sBlo[buf];

    // Fragments (16-bit A layout: lane=M%16; K half by lane>>4).
    bf16x16 ahi[2], alo[2], bhi[4], blo[4];
#pragma unroll
    for (int i = 0; i < 2; ++i) {
      const int m = (wm * 2 + i) * 16 + mrow;
      Pack32 p;
      p.a = *(const uint4*)&pAhi[m * LDSK + khalf * 8];
      p.b = *(const uint4*)&pAhi[m * LDSK + khalf * 8 + 16];
      ahi[i] = __builtin_bit_cast(bf16x16, p);
      p.a = *(const uint4*)&pAlo[m * LDSK + khalf * 8];
      p.b = *(const uint4*)&pAlo[m * LDSK + khalf * 8 + 16];
      alo[i] = __builtin_bit_cast(bf16x16, p);
    }
#pragma unroll
    for (int j = 0; j < 4; ++j) {
      const int n = (wn * 4 + j) * 16 + mrow;   // B col = W row
      Pack32 p;
      p.a = *(const uint4*)&pBhi[n * LDSK + khalf * 8];
      p.b = *(const uint4*)&pBhi[n * LDSK + khalf * 8 + 16];
      bhi[j] = __builtin_bit_cast(bf16x16, p);
      p.a = *(const uint4*)&pBlo[n * LDSK + khalf * 8];
      p.b = *(const uint4*)&pBlo[n * LDSK + khalf * 8 + 16];
      blo[j] = __builtin_bit_cast(bf16x16, p);
    }

#pragma unroll
    for (int i = 0; i < 2; ++i)
#pragma unroll
      for (int j = 0; j < 4; ++j) {
        acc[i][j] = __builtin_amdgcn_wmma_f32_16x16x32_bf16(
            false, ahi[i], false, bhi[j], (short)0, acc[i][j], false, false);
        acc[i][j] = __builtin_amdgcn_wmma_f32_16x16x32_bf16(
            false, ahi[i], false, blo[j], (short)0, acc[i][j], false, false);
        acc[i][j] = __builtin_amdgcn_wmma_f32_16x16x32_bf16(
            false, alo[i], false, bhi[j], (short)0, acc[i][j], false, false);
      }

    wait_async0();     // next-buffer fill complete (issued before compute)
    __syncthreads();
  }

  // Epilogue: D layout col = lane%16, row = (lane>>4)*8 + r.
  const float scale = 0.022097086912079608f;   // 1/sqrt(2048)
#pragma unroll
  for (int i = 0; i < 2; ++i) {
    const int mbase = blockM + (wm * 2 + i) * 16 + khalf * 8;
#pragma unroll
    for (int j = 0; j < 4; ++j) {
      const int col = blockN + (wn * 4 + j) * 16 + mrow;
      const float bb = 0.1f * Bv[col];
#pragma unroll
      for (int r = 0; r < 8; ++r) {
        O[(size_t)(mbase + r) * DOUT + col] = acc[i][j][r] * scale + bb;
      }
    }
  }
}

// ---------------- Fallback: fused convert-in-loop kernel (round-1 version) ----------------
__global__ __launch_bounds__(256)
void linear_bf16x3_fused(const float* __restrict__ X,
                         const float* __restrict__ W,
                         const float* __restrict__ Bv,
                         float* __restrict__ O) {
  __shared__ __align__(16) unsigned short sAhi[BM * LDSK];
  __shared__ __align__(16) unsigned short sAlo[BM * LDSK];
  __shared__ __align__(16) unsigned short sBhi[BN * LDSK];
  __shared__ __align__(16) unsigned short sBlo[BN * LDSK];

  const int tid = threadIdx.x, lane = tid & 31, wave = tid >> 5;
  const int wm = wave >> 1, wn = wave & 1;
  const int mrow = lane & 15, khalf = lane >> 4;
  const int blockM = blockIdx.y * BM, blockN = blockIdx.x * BN;

  floatx8 acc[2][4];
#pragma unroll
  for (int i = 0; i < 2; ++i)
#pragma unroll
    for (int j = 0; j < 4; ++j) acc[i][j] = (floatx8)0.0f;

  for (int k0 = 0; k0 < DIN; k0 += BK) {
    float4 xr[4], wr[4];
#pragma unroll
    for (int p = 0; p < 4; ++p) {
      const int f = tid + p * 256, r = f >> 3, c4 = f & 7;
      xr[p] = ((const float4*)(X + (size_t)(blockM + r) * DIN + k0))[c4];
      wr[p] = ((const float4*)(W + (size_t)(blockN + r) * DIN + k0))[c4];
    }
    __syncthreads();
#pragma unroll
    for (int p = 0; p < 4; ++p) {
      const int f = tid + p * 256, r = f >> 3, c = (f & 7) * 4;
      const float xv[4] = {xr[p].x, xr[p].y, xr[p].z, xr[p].w};
      const float wv[4] = {wr[p].x, wr[p].y, wr[p].z, wr[p].w};
#pragma unroll
      for (int e = 0; e < 4; ++e) {
        const unsigned short hx = f32_to_bf16_rne(xv[e]);
        sAhi[r * LDSK + c + e] = hx;
        sAlo[r * LDSK + c + e] = f32_to_bf16_rne(xv[e] - bf16_to_f32(hx));
        const unsigned short hw = f32_to_bf16_rne(wv[e]);
        sBhi[r * LDSK + c + e] = hw;
        sBlo[r * LDSK + c + e] = f32_to_bf16_rne(wv[e] - bf16_to_f32(hw));
      }
    }
    __syncthreads();

    bf16x16 ahi[2], alo[2], bhi[4], blo[4];
#pragma unroll
    for (int i = 0; i < 2; ++i) {
      const int m = (wm * 2 + i) * 16 + mrow;
      Pack32 p;
      p.a = *(const uint4*)&sAhi[m * LDSK + khalf * 8];
      p.b = *(const uint4*)&sAhi[m * LDSK + khalf * 8 + 16];
      ahi[i] = __builtin_bit_cast(bf16x16, p);
      p.a = *(const uint4*)&sAlo[m * LDSK + khalf * 8];
      p.b = *(const uint4*)&sAlo[m * LDSK + khalf * 8 + 16];
      alo[i] = __builtin_bit_cast(bf16x16, p);
    }
#pragma unroll
    for (int j = 0; j < 4; ++j) {
      const int n = (wn * 4 + j) * 16 + mrow;
      Pack32 p;
      p.a = *(const uint4*)&sBhi[n * LDSK + khalf * 8];
      p.b = *(const uint4*)&sBhi[n * LDSK + khalf * 8 + 16];
      bhi[j] = __builtin_bit_cast(bf16x16, p);
      p.a = *(const uint4*)&sBlo[n * LDSK + khalf * 8];
      p.b = *(const uint4*)&sBlo[n * LDSK + khalf * 8 + 16];
      blo[j] = __builtin_bit_cast(bf16x16, p);
    }
#pragma unroll
    for (int i = 0; i < 2; ++i)
#pragma unroll
      for (int j = 0; j < 4; ++j) {
        acc[i][j] = __builtin_amdgcn_wmma_f32_16x16x32_bf16(
            false, ahi[i], false, bhi[j], (short)0, acc[i][j], false, false);
        acc[i][j] = __builtin_amdgcn_wmma_f32_16x16x32_bf16(
            false, ahi[i], false, blo[j], (short)0, acc[i][j], false, false);
        acc[i][j] = __builtin_amdgcn_wmma_f32_16x16x32_bf16(
            false, alo[i], false, bhi[j], (short)0, acc[i][j], false, false);
      }
  }

  const float scale = 0.022097086912079608f;
#pragma unroll
  for (int i = 0; i < 2; ++i) {
    const int mbase = blockM + (wm * 2 + i) * 16 + khalf * 8;
#pragma unroll
    for (int j = 0; j < 4; ++j) {
      const int col = blockN + (wn * 4 + j) * 16 + mrow;
      const float bb = 0.1f * Bv[col];
#pragma unroll
      for (int r = 0; r < 8; ++r) {
        O[(size_t)(mbase + r) * DOUT + col] = acc[i][j][r] * scale + bb;
      }
    }
  }
}

extern "C" void kernel_launch(void* const* d_in, const int* in_sizes, int n_in,
                              void* d_out, int out_size, void* d_ws, size_t ws_size,
                              hipStream_t stream) {
  const float* X  = (const float*)d_in[0];   // (16384, 2048)
  const float* W  = (const float*)d_in[1];   // (2048, 2048)
  const float* Bv = (const float*)d_in[2];   // (1, 2048)
  float* O        = (float*)d_out;           // (16384, 2048)

  const size_t X_ELEMS = (size_t)NROWS * DIN;
  const size_t W_ELEMS = (size_t)DOUT * DIN;
  const size_t need = (X_ELEMS + W_ELEMS) * 2 /*planes*/ * 2 /*bytes*/;

  dim3 grid(DOUT / BN, NROWS / BM);   // 16 x 128
  dim3 block(256);

  if (ws_size >= need) {
    unsigned short* Xhi = (unsigned short*)d_ws;
    unsigned short* Xlo = Xhi + X_ELEMS;
    unsigned short* Whi = Xlo + X_ELEMS;
    unsigned short* Wlo = Whi + W_ELEMS;
    const int xn4 = (int)(X_ELEMS / 4), wn4 = (int)(W_ELEMS / 4);
    hipLaunchKernelGGL(split_planes, dim3((xn4 + 255) / 256), block, 0, stream,
                       X, Xhi, Xlo, xn4);
    hipLaunchKernelGGL(split_planes, dim3((wn4 + 255) / 256), block, 0, stream,
                       W, Whi, Wlo, wn4);
    hipLaunchKernelGGL(gemm_planes_wmma, grid, block, 0, stream,
                       Xhi, Xlo, Whi, Wlo, Bv, O);
  } else {
    hipLaunchKernelGGL(linear_bf16x3_fused, grid, block, 0, stream, X, W, Bv, O);
  }
}
